// Attn_35407710389017
// MI455X (gfx1250) — matrix operand order repaired and checked by reading, not compiled, not run on hardware
//
#include <hip/hip_runtime.h>

// MI455X / gfx1250 flash-attention pipeline.
// All GEMMs use v_wmma_f32_16x16x32_f16 (f16 in, f32 accumulate).
// Flash-attention K/V tiles double-buffered via the Tensor Data Mover
// (tensor_load_to_lds + s_wait_tensorcnt) when available; manual staging else.
// GEMM k-loops are software-pipelined: next tile's global loads issued into
// registers while the current tile computes from LDS.
// Workspace layout (f16): qbuf | kbuf | vT | obuf, each 25,165,824 elems (~192MB).

typedef __attribute__((ext_vector_type(16))) _Float16 v16h;
typedef __attribute__((ext_vector_type(8)))  float    v8f;

union FragU { v16h v; unsigned int u[8]; _Float16 h[16]; };

__device__ __forceinline__ v8f wmma32(v16h a, v16h b, v8f c) {
  // D = A(16x32 f16) * B(32x16 f16) + C(16x16 f32)
  return __builtin_amdgcn_wmma_f32_16x16x32_f16(false, a, false, b, (short)0, c,
                                                false, false);
}

// A-matrix fragment (16x32 f16, row-major, stride ld halves).
// ISA layout: lanes 0-15 row M=lane, VGPR0-3 = K{0,1..6,7}, VGPR4-7 = K{16..23};
// lanes 16-31 same rows, K offset +8.
__device__ __forceinline__ v16h frag_a(const _Float16* p, int row0, int k0, int ld) {
  const int lane = threadIdx.x & 31;
  const int hf = lane >> 4, rr = lane & 15;
  FragU f;
#pragma unroll
  for (int j = 0; j < 8; ++j) {
    const int k = ((j < 4) ? (2 * j) : (16 + 2 * (j - 4))) + 8 * hf;
    f.u[j] = *(const unsigned int*)(p + (size_t)(row0 + rr) * ld + k0 + k);
  }
  return f.v;
}

// B-matrix fragment (32x16 f16). Storage p is [n][k] row-major, stride ld halves.
// ISA layout: lane r -> column N=r; VGPR0-3 = K{0..7}, VGPR4-7 = K{16..23}, +8 for lanes 16-31.
__device__ __forceinline__ v16h frag_b(const _Float16* p, int n0, int k0, int ld) {
  const int lane = threadIdx.x & 31;
  const int hf = lane >> 4, rr = lane & 15;
  FragU f;
#pragma unroll
  for (int j = 0; j < 8; ++j) {
    const int k = ((j < 4) ? (2 * j) : (16 + 2 * (j - 4))) + 8 * hf;
    f.u[j] = *(const unsigned int*)(p + (size_t)(n0 + rr) * ld + k0 + k);
  }
  return f.v;
}

// ---------------------------------------------------------------------------
// Tensor Data Mover: 2D tile load global -> LDS with padding so that a
// 64-halves (128B) tile row lands at a 68-half (136B) LDS stride.
// pad_interval code 4 = every 32 dwords, pad_amount code 1 = 2 dwords.
// ---------------------------------------------------------------------------
#if __has_builtin(__builtin_amdgcn_tensor_load_to_lds) && \
    __has_builtin(__builtin_amdgcn_s_wait_tensorcnt)
#define USE_TDM 1
typedef __attribute__((ext_vector_type(4))) unsigned int u32x4;
typedef __attribute__((ext_vector_type(4))) int i32x4;
typedef __attribute__((ext_vector_type(8))) int i32x8;

__device__ __forceinline__ void tdm_load_2d_pad68(
    const _Float16* gsrc, unsigned lds_byte_off, unsigned rows, unsigned rowlen,
    unsigned row_stride_elems) {
  const unsigned long long ga = (unsigned long long)(const void*)gsrc;
  u32x4 g0;
  g0.x = 1u;                                   // count=1 user descriptor
  g0.y = lds_byte_off;                         // lds_addr[63:32]
  g0.z = (unsigned)(ga & 0xffffffffu);         // global_addr low
  g0.w = (unsigned)((ga >> 32) & 0x01ffffffu)  // global_addr high bits [120:64]
         | (2u << 30);                         // type=2 ("image")
  i32x8 g1;
  g1[0] = (1 << 16)      // data_size = 2 bytes
        | (1 << 20)      // pad_enable
        | (4 << 22)      // pad_interval: 32 dwords (=128B tile row)
        | (1 << 25);     // pad_amount: 2 dwords (=4 halves) -> stride 68 halves
  g1[1] = (int)(rowlen << 16);                   // tensor_dim0 low16  (bits 63:48)
  g1[2] = (int)((rowlen >> 16) | (rows << 16));  // dim0 hi16 | tensor_dim1 lo16
  g1[3] = (int)((rows >> 16) | (rowlen << 16));  // dim1 hi16 | tile_dim0
  g1[4] = (int)rows;                             // tile_dim1 (tile_dim2=0)
  g1[5] = (int)row_stride_elems;                 // tensor_dim0_stride low32
  g1[6] = 0;                                     // stride hi16 | dim1_stride lo16
  g1[7] = 0;
  const i32x4 z4 = {0, 0, 0, 0};
#if __clang_major__ >= 23
  const i32x8 z8 = {0, 0, 0, 0, 0, 0, 0, 0};
  __builtin_amdgcn_tensor_load_to_lds(g0, g1, z4, z4, z8, 0);
#else
  __builtin_amdgcn_tensor_load_to_lds(g0, g1, z4, z4, 0);
#endif
}
#endif

// ---------------------------------------------------------------------------
// Kernel 1: qkv = x @ Wqkv^T  (M=65536, N=1152, K=384), f16 compute / f32 acc.
// Software-pipelined k-loop; epilogue scatters Q,K -> (B,NH,T,HD) f16 and
// V -> transposed (B,NH,HD,T) f16.
// ---------------------------------------------------------------------------
__global__ __launch_bounds__(256) void qkv_gemm_pack(
    const float* __restrict__ x, const float* __restrict__ Wqkv,
    _Float16* __restrict__ qb, _Float16* __restrict__ kb,
    _Float16* __restrict__ vtb) {
  __shared__ _Float16 As[128 * 34];  // 128 rows x 32 k, stride 34 (odd dword stride)
  __shared__ _Float16 Bs[64 * 34];   // 64 n  x 32 k
  const int mblk = blockIdx.x / 18, nblk = blockIdx.x % 18;
  const int m0 = mblk * 128, n0 = nblk * 64;
  const int tid = threadIdx.x;
  const int wave = tid >> 5, lane = tid & 31;
  const int hf = lane >> 4, rr = lane & 15;
  const int wm = (wave >> 1) * 32, wn = (wave & 1) * 32;

  v8f acc[2][2] = {};
  float2 ra[8], rb[4];  // in-flight next tile (load phase fully before store)

  auto load_tiles = [&](int k0) {
#pragma unroll
    for (int it = 0; it < 8; ++it) {
      const int idx = it * 256 + tid;
      const int row = idx >> 4, seg = (idx & 15) * 2;
      ra[it] = *(const float2*)(x + (size_t)(m0 + row) * 384 + k0 + seg);
    }
#pragma unroll
    for (int it = 0; it < 4; ++it) {
      const int idx = it * 256 + tid;
      const int row = idx >> 4, seg = (idx & 15) * 2;
      rb[it] = *(const float2*)(Wqkv + (size_t)(n0 + row) * 384 + k0 + seg);
    }
  };
  auto store_tiles = [&]() {
#pragma unroll
    for (int it = 0; it < 8; ++it) {
      const int idx = it * 256 + tid;
      const int row = idx >> 4, seg = (idx & 15) * 2;
      union { _Float16 h[2]; unsigned int u; } pk;
      pk.h[0] = (_Float16)ra[it].x; pk.h[1] = (_Float16)ra[it].y;
      *(unsigned int*)&As[row * 34 + seg] = pk.u;
    }
#pragma unroll
    for (int it = 0; it < 4; ++it) {
      const int idx = it * 256 + tid;
      const int row = idx >> 4, seg = (idx & 15) * 2;
      union { _Float16 h[2]; unsigned int u; } pk;
      pk.h[0] = (_Float16)rb[it].x; pk.h[1] = (_Float16)rb[it].y;
      *(unsigned int*)&Bs[row * 34 + seg] = pk.u;
    }
  };

  load_tiles(0);
  for (int k0 = 0; k0 < 384; k0 += 32) {
    store_tiles();
    __syncthreads();
    if (k0 + 32 < 384) load_tiles(k0 + 32);  // overlap next fetch with compute

    const v16h a0 = frag_a(As, wm, 0, 34);
    const v16h a1 = frag_a(As, wm + 16, 0, 34);
    const v16h b0 = frag_b(Bs, wn, 0, 34);
    const v16h b1 = frag_b(Bs, wn + 16, 0, 34);
    acc[0][0] = wmma32(a0, b0, acc[0][0]);
    acc[0][1] = wmma32(a0, b1, acc[0][1]);
    acc[1][0] = wmma32(a1, b0, acc[1][0]);
    acc[1][1] = wmma32(a1, b1, acc[1][1]);
    __syncthreads();
  }

  // Scatter epilogue. Each 16-wide N-subtile lies entirely inside one head of
  // one of q/k/v (16 | 64 | 384), so which/h/e_base are wave-uniform scalars.
#pragma unroll
  for (int jn = 0; jn < 2; ++jn) {
    const int nbase = n0 + wn + jn * 16;
    const int which = nbase / 384;
    const int rem = nbase % 384;
    const int h = rem >> 6;
    const int ebase = rem & 63;
#pragma unroll
    for (int i = 0; i < 2; ++i)
#pragma unroll
      for (int jj = 0; jj < 8; ++jj) {
        const int m = m0 + wm + i * 16 + jj + 8 * hf;
        const int b = m >> 9, t = m & 511;
        const size_t bh = (size_t)(b * 6 + h);
        const _Float16 val = (_Float16)acc[i][jn][jj];
        if (which == 0)
          qb[(bh * 512 + t) * 64 + ebase + rr] = val;
        else if (which == 1)
          kb[(bh * 512 + t) * 64 + ebase + rr] = val;
        else
          vtb[(bh * 64 + ebase + rr) * 512 + t] = val;
      }
  }
}

// ---------------------------------------------------------------------------
// Kernel 2: in-place RoPE on Q,K first 16 dims (pairs e <-> e+8).
// ---------------------------------------------------------------------------
__global__ __launch_bounds__(256) void rope_qk(
    _Float16* __restrict__ qb, _Float16* __restrict__ kb,
    const float* __restrict__ cosb, const float* __restrict__ sinb) {
  const int gid = blockIdx.x * 256 + threadIdx.x;  // = bh*512 + t, 0..393215
  const int t = gid & 511;
  const size_t base = (size_t)gid * 64;
#pragma unroll
  for (int f = 0; f < 8; ++f) {
    const float c = cosb[t * 8 + f], s = sinb[t * 8 + f];
    const float q1 = (float)qb[base + f], q2 = (float)qb[base + 8 + f];
    qb[base + f]     = (_Float16)(q1 * c - q2 * s);
    qb[base + 8 + f] = (_Float16)(q2 * c + q1 * s);
    const float k1 = (float)kb[base + f], k2 = (float)kb[base + 8 + f];
    kb[base + f]     = (_Float16)(k1 * c - k2 * s);
    kb[base + 8 + f] = (_Float16)(k2 * c + k1 * s);
  }
}

// ---------------------------------------------------------------------------
// Kernel 3: causal flash attention. One block per (b,h,query-block of 64).
// 4 waves x 16 query rows; 64-key blocks double-buffered in LDS: wave 0 issues
// the next block's TDM pair while the current block computes, then waits
// s_wait_tensorcnt 2 (next pair may stay in flight; TENSORcnt retires in order).
// ---------------------------------------------------------------------------
__global__ __launch_bounds__(128) void flash_attn(
    const _Float16* __restrict__ qb, const _Float16* __restrict__ kb,
    const _Float16* __restrict__ vtb, _Float16* __restrict__ ob) {
  __shared__ _Float16 Kl[2][64 * 68];    // [buf][key][e]
  __shared__ _Float16 Vl[2][64 * 68];    // [buf][e][key]  (V stored transposed)
  __shared__ _Float16 Pl[4 * 16 * 68];   // per-wave P staging (C-frag -> A-frag)

  const int bh = blockIdx.x >> 3;
  const int qbk = blockIdx.x & 7;
  const int tid = threadIdx.x;
  const int wave = tid >> 5, lane = tid & 31;
  const int hf = lane >> 4, rr = lane & 15;
  const int q0 = qbk * 64 + wave * 16;
  const int b = bh / 6, h = bh % 6;

  // Q fragments for this wave's 16 rows, resident across the whole loop.
  const _Float16* qbase = qb + (size_t)bh * 512 * 64;
  const v16h qa0 = frag_a(qbase, q0, 0, 64);
  const v16h qa1 = frag_a(qbase, q0, 32, 64);

  v8f oac[4] = {};
  float mrow[8], lrow[8];
#pragma unroll
  for (int jj = 0; jj < 8; ++jj) { mrow[jj] = -1e30f; lrow[jj] = 0.f; }

  _Float16* Plw = &Pl[wave * 16 * 68];

#ifdef USE_TDM
  auto issue_kv = [&](int blk, int bufi) {
    tdm_load_2d_pad68(kb + ((size_t)bh * 512 + blk * 64) * 64,
                      (unsigned)(size_t)(const void*)Kl[bufi], 64, 64, 64);
    tdm_load_2d_pad68(vtb + ((size_t)bh * 64) * 512 + blk * 64,
                      (unsigned)(size_t)(const void*)Vl[bufi], 64, 64, 512);
  };
  if (wave == 0) issue_kv(0, 0);
#endif

  int cur = 0;
  for (int kbi = 0; kbi <= qbk; ++kbi) {
#ifdef USE_TDM
    if (wave == 0) {
      if (kbi < qbk) {
        issue_kv(kbi + 1, cur ^ 1);            // prefetch next block
        __builtin_amdgcn_s_wait_tensorcnt(2);  // current pair complete
      } else {
        __builtin_amdgcn_s_wait_tensorcnt(0);
      }
    }
    __syncthreads();
#else
    {
      uint2 kr[8], vr[8];
      const uint2* kg = (const uint2*)(kb + ((size_t)bh * 512 + kbi * 64) * 64);
#pragma unroll
      for (int it = 0; it < 8; ++it) kr[it] = kg[it * 128 + tid];
#pragma unroll
      for (int it = 0; it < 8; ++it) {
        const int idx = it * 128 + tid;
        const int e = idx >> 4, seg = idx & 15;
        vr[it] = ((const uint2*)(vtb + ((size_t)bh * 64 + e) * 512 +
                                 kbi * 64))[seg];
      }
#pragma unroll
      for (int it = 0; it < 8; ++it) {
        const int idx = it * 128 + tid;
        const int key = idx >> 4, seg = idx & 15;
        *(uint2*)&Kl[cur][key * 68 + seg * 4] = kr[it];
      }
#pragma unroll
      for (int it = 0; it < 8; ++it) {
        const int idx = it * 128 + tid;
        const int e = idx >> 4, seg = idx & 15;
        *(uint2*)&Vl[cur][e * 68 + seg * 4] = vr[it];
      }
    }
    __syncthreads();
#endif

    // Scores: 16x64 via 4 tiles x 2 WMMAs (HD=64 = 2 x K32)
    v8f s[4];
#pragma unroll
    for (int nt = 0; nt < 4; ++nt) {
      v8f a = {};
      a = wmma32(qa0, frag_b(Kl[cur], nt * 16, 0, 68), a);
      a = wmma32(qa1, frag_b(Kl[cur], nt * 16, 32, 68), a);
      s[nt] = a;
    }
    // Scale + causal mask (col = key in lane, row = q in VGPR index)
#pragma unroll
    for (int nt = 0; nt < 4; ++nt)
#pragma unroll
      for (int jj = 0; jj < 8; ++jj) {
        const int key = kbi * 64 + nt * 16 + rr;
        const int qq = q0 + jj + 8 * hf;
        const float v = s[nt][jj] * 0.125f;  // 1/sqrt(64)
        s[nt][jj] = (key <= qq) ? v : -1e30f;
      }
    // Online softmax: row stats live across the 16 lanes of each half-wave.
    float alpha[8];
#pragma unroll
    for (int jj = 0; jj < 8; ++jj) {
      float pm = fmaxf(fmaxf(s[0][jj], s[1][jj]), fmaxf(s[2][jj], s[3][jj]));
      pm = fmaxf(pm, __shfl_xor(pm, 1));
      pm = fmaxf(pm, __shfl_xor(pm, 2));
      pm = fmaxf(pm, __shfl_xor(pm, 4));
      pm = fmaxf(pm, __shfl_xor(pm, 8));
      const float mn = fmaxf(mrow[jj], pm);
      alpha[jj] = __expf(mrow[jj] - mn);
      mrow[jj] = mn;
      float rs = 0.f;
#pragma unroll
      for (int nt = 0; nt < 4; ++nt) {
        const float p = __expf(s[nt][jj] - mn);
        s[nt][jj] = p;
        rs += p;
      }
      rs += __shfl_xor(rs, 1);
      rs += __shfl_xor(rs, 2);
      rs += __shfl_xor(rs, 4);
      rs += __shfl_xor(rs, 8);
      lrow[jj] = lrow[jj] * alpha[jj] + rs;
    }
    // Re-layout P (C-frag) -> A-frag via per-wave LDS; rescale O accumulators.
#pragma unroll
    for (int nt = 0; nt < 4; ++nt)
#pragma unroll
      for (int jj = 0; jj < 8; ++jj)
        Plw[(jj + 8 * hf) * 68 + nt * 16 + rr] = (_Float16)s[nt][jj];
#pragma unroll
    for (int et = 0; et < 4; ++et)
#pragma unroll
      for (int jj = 0; jj < 8; ++jj) oac[et][jj] *= alpha[jj];

    const v16h ap0 = frag_a(Plw, 0, 0, 68);
    const v16h ap1 = frag_a(Plw, 0, 32, 68);
#pragma unroll
    for (int et = 0; et < 4; ++et) {
      oac[et] = wmma32(ap0, frag_b(Vl[cur], et * 16, 0, 68), oac[et]);
      oac[et] = wmma32(ap1, frag_b(Vl[cur], et * 16, 32, 68), oac[et]);
    }
    __syncthreads();  // all reads of buf[cur] done before it is overwritten
    cur ^= 1;
  }

  // Epilogue: O = acc / l, store f16 in (B,T,C) layout for the out-projection.
#pragma unroll
  for (int et = 0; et < 4; ++et)
#pragma unroll
    for (int jj = 0; jj < 8; ++jj) {
      const int t = q0 + jj + 8 * hf;
      const int c = h * 64 + et * 16 + rr;
      const float ov = oac[et][jj] / lrow[jj];
      ob[((size_t)b * 512 + t) * 384 + c] = (_Float16)ov;
    }
}

// ---------------------------------------------------------------------------
// Kernel 4: out = o @ Wout^T  (M=65536, N=384, K=384), f32 output.
// Software-pipelined k-loop like kernel 1.
// ---------------------------------------------------------------------------
__global__ __launch_bounds__(256) void out_proj(
    const _Float16* __restrict__ ob, const float* __restrict__ Wout,
    float* __restrict__ out) {
  __shared__ _Float16 As[128 * 34];
  __shared__ _Float16 Bs[64 * 34];
  const int mblk = blockIdx.x / 6, nblk = blockIdx.x % 6;
  const int m0 = mblk * 128, n0 = nblk * 64;
  const int tid = threadIdx.x;
  const int wave = tid >> 5, lane = tid & 31;
  const int hf = lane >> 4, rr = lane & 15;
  const int wm = (wave >> 1) * 32, wn = (wave & 1) * 32;

  v8f acc[2][2] = {};
  unsigned int ra[8];
  float2 rb[4];

  auto load_tiles = [&](int k0) {
#pragma unroll
    for (int it = 0; it < 8; ++it) {
      const int idx = it * 256 + tid;
      const int row = idx >> 4, seg = (idx & 15) * 2;
      ra[it] = *(const unsigned int*)(ob + (size_t)(m0 + row) * 384 + k0 + seg);
    }
#pragma unroll
    for (int it = 0; it < 4; ++it) {
      const int idx = it * 256 + tid;
      const int row = idx >> 4, seg = (idx & 15) * 2;
      rb[it] = *(const float2*)(Wout + (size_t)(n0 + row) * 384 + k0 + seg);
    }
  };
  auto store_tiles = [&]() {
#pragma unroll
    for (int it = 0; it < 8; ++it) {
      const int idx = it * 256 + tid;
      const int row = idx >> 4, seg = (idx & 15) * 2;
      *(unsigned int*)&As[row * 34 + seg] = ra[it];
    }
#pragma unroll
    for (int it = 0; it < 4; ++it) {
      const int idx = it * 256 + tid;
      const int row = idx >> 4, seg = (idx & 15) * 2;
      union { _Float16 h[2]; unsigned int u; } pk;
      pk.h[0] = (_Float16)rb[it].x; pk.h[1] = (_Float16)rb[it].y;
      *(unsigned int*)&Bs[row * 34 + seg] = pk.u;
    }
  };

  load_tiles(0);
  for (int k0 = 0; k0 < 384; k0 += 32) {
    store_tiles();
    __syncthreads();
    if (k0 + 32 < 384) load_tiles(k0 + 32);

    const v16h a0 = frag_a(As, wm, 0, 34);
    const v16h a1 = frag_a(As, wm + 16, 0, 34);
    const v16h b0 = frag_b(Bs, wn, 0, 34);
    const v16h b1 = frag_b(Bs, wn + 16, 0, 34);
    acc[0][0] = wmma32(a0, b0, acc[0][0]);
    acc[0][1] = wmma32(a0, b1, acc[0][1]);
    acc[1][0] = wmma32(a1, b0, acc[1][0]);
    acc[1][1] = wmma32(a1, b1, acc[1][1]);
    __syncthreads();
  }

#pragma unroll
  for (int i = 0; i < 2; ++i)
#pragma unroll
    for (int jn = 0; jn < 2; ++jn)
#pragma unroll
      for (int jj = 0; jj < 8; ++jj) {
        const int m = m0 + wm + i * 16 + jj + 8 * hf;
        const int n = n0 + wn + jn * 16 + rr;
        out[(size_t)m * 384 + n] = acc[i][jn][jj];
      }
}

// ---------------------------------------------------------------------------
extern "C" void kernel_launch(void* const* d_in, const int* in_sizes, int n_in,
                              void* d_out, int out_size, void* d_ws,
                              size_t ws_size, hipStream_t stream) {
  const float* x    = (const float*)d_in[0];
  const float* Wqkv = (const float*)d_in[1];
  const float* Wout = (const float*)d_in[2];
  const float* cosb = (const float*)d_in[3];
  const float* sinb = (const float*)d_in[4];
  float* out = (float*)d_out;

  const size_t NBH = (size_t)128 * 6 * 512 * 64;  // 25,165,824 elems per buffer
  _Float16* qb  = (_Float16*)d_ws;
  _Float16* kb  = qb + NBH;
  _Float16* vtb = kb + NBH;
  _Float16* ob  = vtb + NBH;   // (B,T,C) f16, also 25,165,824 elems

  qkv_gemm_pack<<<512 * 18, 256, 0, stream>>>(x, Wqkv, qb, kb, vtb);
  rope_qk<<<1536, 256, 0, stream>>>(qb, kb, cosb, sinb);
  flash_attn<<<128 * 6 * 8, 128, 0, stream>>>(qb, kb, vtb, ob);
  out_proj<<<512 * 6, 256, 0, stream>>>(ob, Wout, out);
}
